// MultiHeadAttention_19215683682474
// MI455X (gfx1250) — matrix operand order repaired and checked by reading, not compiled, not run on hardware
//
#include <hip/hip_runtime.h>

// ---------------- types / helpers ----------------
typedef __bf16 v16bf __attribute__((ext_vector_type(16)));
typedef float  v8f   __attribute__((ext_vector_type(8)));

union U16 {
    v16bf v;
    unsigned int u[8];
};

__device__ __forceinline__ unsigned short f2bf(float f) {
    unsigned int u = __float_as_uint(f);
    unsigned int r = u + 0x7FFFu + ((u >> 16) & 1u);   // round-to-nearest-even
    return (unsigned short)(r >> 16);
}

__device__ __forceinline__ v8f vzero8() {
    v8f z;
#pragma unroll
    for (int i = 0; i < 8; i++) z[i] = 0.0f;
    return z;
}

__device__ __forceinline__ float rowmax16(float v) {
#pragma unroll
    for (int m = 1; m < 16; m <<= 1) v = fmaxf(v, __shfl_xor(v, m, 16));
    return v;
}
__device__ __forceinline__ float rowsum16(float v) {
#pragma unroll
    for (int m = 1; m < 16; m <<= 1) v += __shfl_xor(v, m, 16);
    return v;
}

// async global->LDS dword copy (gfx1250 ASYNCcnt path) with safe fallback
__device__ __forceinline__ void async_g2l_b32(unsigned int* ldsdst,
                                              const unsigned int* gsrc) {
#if __has_builtin(__builtin_amdgcn_global_load_async_to_lds_b32)
    __builtin_amdgcn_global_load_async_to_lds_b32(
        (__attribute__((address_space(1))) int*)gsrc,
        (__attribute__((address_space(3))) int*)ldsdst, 0, 0);
#else
    *ldsdst = *gsrc;
#endif
}

__device__ __forceinline__ void wait_async0() {
    asm volatile("s_wait_asynccnt 0x0" ::: "memory");
}

// constants
#define BB 8
#define SS 1024
#define DD 1024
#define HH 16
#define DKV 64

// ---------------- LayerNorm + bf16 cast ----------------
__global__ __launch_bounds__(256) void ln_cast_kernel(
    const float* __restrict__ x, const float* __restrict__ g,
    const float* __restrict__ bta, unsigned short* __restrict__ xn)
{
    __shared__ float red[256];
    const int row = blockIdx.x;
    const int tid = threadIdx.x;
    const float* xr = x + (size_t)row * DD;

    float v[4];
    float s = 0.0f;
#pragma unroll
    for (int i = 0; i < 4; i++) { v[i] = xr[tid + i * 256]; s += v[i]; }
    red[tid] = s;
    __syncthreads();
    for (int st = 128; st > 0; st >>= 1) {
        if (tid < st) red[tid] += red[tid + st];
        __syncthreads();
    }
    const float mu = red[0] * (1.0f / DD);
    __syncthreads();

    float q = 0.0f;
#pragma unroll
    for (int i = 0; i < 4; i++) { float d = v[i] - mu; q += d * d; }
    red[tid] = q;
    __syncthreads();
    for (int st = 128; st > 0; st >>= 1) {
        if (tid < st) red[tid] += red[tid + st];
        __syncthreads();
    }
    const float rstd = rsqrtf(red[0] * (1.0f / DD) + 1e-5f);

    unsigned short* xo = xn + (size_t)row * DD;
#pragma unroll
    for (int i = 0; i < 4; i++) {
        int c = tid + i * 256;
        xo[c] = f2bf((v[i] - mu) * rstd * g[c] + bta[c]);
    }
}

// ---------------- f32 -> bf16 cast ----------------
__global__ __launch_bounds__(256) void cast_kernel(
    const float* __restrict__ src, unsigned short* __restrict__ dst, int n)
{
    int i = blockIdx.x * 256 + threadIdx.x;
    if (i < n) dst[i] = f2bf(src[i]);
}

// ---------------- bf16 WMMA GEMM ----------------
// C[M,N] = A[M,K] @ W[K,N] + bias (+ residual if FINAL)
// block = 256 threads (8 waves); macro tile 128(M) x 64(N); K-chunks of 32
template <bool FINAL>
__global__ __launch_bounds__(256) void gemm_bf16_kernel(
    const unsigned short* __restrict__ A,    // [M,K] bf16
    const unsigned short* __restrict__ W,    // [K,N] bf16
    const float* __restrict__ bias,          // [N]
    const float* __restrict__ residual,      // [M,N] f32 (FINAL only)
    unsigned short* __restrict__ outb,       // [M,N] bf16 (!FINAL)
    float* __restrict__ outf,                // [M,N] f32  (FINAL)
    int M, int N, int K)
{
    // padded strides (odd dword counts) to avoid LDS bank conflicts
    __shared__ unsigned int aAs[128 * 17];   // A tile: [row][feat-pair], stride 17
    __shared__ unsigned int bBs_u[64 * 17];  // B^T tile: [n][kk-pair], stride 17
    unsigned short* bBs = (unsigned short*)bBs_u;

    const int tid  = threadIdx.x;
    const int w    = tid >> 5;
    const int lane = tid & 31;
    const int half = lane >> 4;
    const int lm   = lane & 15;
    const int mbase = blockIdx.y * 128;
    const int nbase = blockIdx.x * 64;

    v8f acc[4];
#pragma unroll
    for (int a = 0; a < 4; a++) acc[a] = vzero8();

    const unsigned int* Au = (const unsigned int*)A;
    const unsigned int* Wu = (const unsigned int*)W;

    for (int k0 = 0; k0 < K; k0 += 32) {
        // stage A: 128 rows x 32 feats = 2048 dwords (8/thread), coalesced
#pragma unroll
        for (int i = 0; i < 8; i++) {
            int d = tid + i * 256;
            int row = d >> 4, fp = d & 15;
            aAs[row * 17 + fp] = Au[(size_t)(mbase + row) * (K >> 1) + (k0 >> 1) + fp];
        }
        // stage B transposed into [n][kk]: 32 rows x 64 cols = 1024 dwords
#pragma unroll
        for (int i = 0; i < 4; i++) {
            int d = tid + i * 256;
            int kk = d >> 5, np = d & 31;
            unsigned int wv = Wu[(size_t)(k0 + kk) * (N >> 1) + (nbase >> 1) + np];
            int n0 = np * 2;
            bBs[n0 * 34 + kk]       = (unsigned short)(wv & 0xFFFFu);
            bBs[(n0 + 1) * 34 + kk] = (unsigned short)(wv >> 16);
        }
        __syncthreads();

        // A fragment: lane = row M (= w*16+lm), VGPR r -> K per ISA layout
        U16 af;
#pragma unroll
        for (int r = 0; r < 8; r++) {
            int kf = ((r < 4) ? 0 : 16) + (half ? 8 : 0) + 2 * (r & 3);
            af.u[r] = aAs[(w * 16 + lm) * 17 + (kf >> 1)];
        }
#pragma unroll
        for (int a = 0; a < 4; a++) {
            U16 bf_;
#pragma unroll
            for (int r = 0; r < 8; r++)  // lane = col N, VGPR r -> K = (r<4?0:16)+half*8+2(r&3)
                bf_.u[r] = bBs_u[(a * 16 + lm) * 17 + ((r < 4) ? 0 : 8) + half * 4 + (r & 3)];
            acc[a] = __builtin_amdgcn_wmma_f32_16x16x32_bf16(
                false, af.v, false, bf_.v, (short)0, acc[a], false, false);
        }
        __syncthreads();
    }

    // epilogue: C layout lane -> N = lm, VGPR r -> M = r + 8*half
#pragma unroll
    for (int a = 0; a < 4; a++) {
        int col = nbase + a * 16 + lm;
        float bv = bias[col];
#pragma unroll
        for (int r = 0; r < 8; r++) {
            int row = mbase + w * 16 + r + 8 * half;
            float val = acc[a][r] + bv;
            if (FINAL) {
                outf[(size_t)row * N + col] = val + residual[(size_t)row * N + col];
            } else {
                outb[(size_t)row * N + col] = f2bf(val);
            }
        }
    }
}

// ---------------- flash attention (bf16 WMMA, f32 softmax) ----------------
// grid (S/128, H, B), block 256 (8 waves). Each wave: 16 query rows.
// Double-buffered LDS K/V tiles; K staged via async global->LDS (ASYNCcnt),
// V register-pipelined + transposed into LDS. Compute overlaps next-tile loads.
__global__ __launch_bounds__(256) void attn_kernel(
    const unsigned short* __restrict__ q,    // [B,S,H*DK] bf16
    const unsigned short* __restrict__ kv,   // [B,S,2*H*DV] bf16 (k | v)
    const int* __restrict__ lengths,
    unsigned short* __restrict__ ao)         // [B,S,H*DV] bf16
{
    __shared__ unsigned int Ks_u[2][64 * 33];  // K tile [key][feat-pair], stride 33
    __shared__ unsigned int Vs_u[2][64 * 34];  // V^T tile [dv][key-pair], stride 34
    __shared__ float p_lds[8][16 * 65];        // per-wave C->A transpose, stride 65

    const int tid  = threadIdx.x;
    const int w    = tid >> 5;
    const int lane = tid & 31;
    const int half = lane >> 4;
    const int lm   = lane & 15;
    const int b = blockIdx.z, h = blockIdx.y;
    const int qbase = blockIdx.x * 128 + w * 16;
    const int len = lengths[b];
    const float scale = 0.125f;                // 1/sqrt(64)

    const unsigned int* qu  = (const unsigned int*)q;
    const unsigned int* kvu = (const unsigned int*)kv;

    // per-thread staging coordinates: d = tid + i*256 -> key = d>>5, dp = d&31
    // K tile: 64 keys x 32 dword feat-pairs; V identical then transposed.

    // Q fragments (features 0-31, 32-63)
    U16 aq[2];
    {
        int qrow = qbase + lm;
        size_t base = (size_t)(b * SS + qrow) * (HH * DKV / 2) + h * (DKV / 2);
#pragma unroll
        for (int fc = 0; fc < 2; fc++)
#pragma unroll
            for (int r = 0; r < 8; r++) {
                int kf = fc * 32 + ((r < 4) ? 0 : 16) + (half ? 8 : 0) + 2 * (r & 3);
                aq[fc].u[r] = qu[base + (kf >> 1)];
            }
    }

    v8f o[4];
#pragma unroll
    for (int a = 0; a < 4; a++) o[a] = vzero8();
    float mrow[8], lrow[8];
#pragma unroll
    for (int r = 0; r < 8; r++) { mrow[r] = -1e30f; lrow[r] = 0.0f; }

    const int nkb = (len + 63) >> 6;
    unsigned int vreg[8];

    // ---- prologue: stage tile 0 into buffer 0 ----
#pragma unroll
    for (int i = 0; i < 8; i++) {
        int d = tid + i * 256;
        int key = d >> 5, dp = d & 31;
        size_t rowu = (size_t)(b * SS + key) * (2 * HH * DKV / 2);
        async_g2l_b32(&Ks_u[0][key * 33 + dp], &kvu[rowu + h * (DKV / 2) + dp]);
        vreg[i] = kvu[rowu + (HH * DKV / 2) + h * (DKV / 2) + dp];
    }
    {
        unsigned short* Vs_s = (unsigned short*)Vs_u[0];
#pragma unroll
        for (int i = 0; i < 8; i++) {
            int d = tid + i * 256;
            int key = d >> 5, dp = d & 31;
            int dv0 = dp * 2;
            Vs_s[dv0 * 68 + key]       = (unsigned short)(vreg[i] & 0xFFFFu);
            Vs_s[(dv0 + 1) * 68 + key] = (unsigned short)(vreg[i] >> 16);
        }
    }
    wait_async0();
    __syncthreads();

    for (int kb = 0; kb < nkb; kb++) {
        const int kbase = kb * 64;
        const int buf = kb & 1;
        const bool more = (kb + 1 < nkb);

        // ---- issue next tile's loads (overlap with compute below) ----
        if (more) {
#pragma unroll
            for (int i = 0; i < 8; i++) {
                int d = tid + i * 256;
                int key = d >> 5, dp = d & 31;
                size_t rowu = (size_t)(b * SS + kbase + 64 + key) * (2 * HH * DKV / 2);
                async_g2l_b32(&Ks_u[buf ^ 1][key * 33 + dp],
                              &kvu[rowu + h * (DKV / 2) + dp]);
                vreg[i] = kvu[rowu + (HH * DKV / 2) + h * (DKV / 2) + dp];
            }
        }

        // ---- S = Q @ K^T : four 16-key N-subtiles, K-dim 64 split 2x32 ----
        v8f s[4];
#pragma unroll
        for (int a = 0; a < 4; a++) {
            int key_l = a * 16 + lm;           // lane = B-matrix column N = key
            U16 bk0, bk1;
#pragma unroll
            for (int r = 0; r < 8; r++) {
                int fp = ((r < 4) ? 0 : 8) + half * 4 + (r & 3);         // K feature pair per B layout
                bk0.u[r] = Ks_u[buf][key_l * 33 + fp];        // feats 0..31
                bk1.u[r] = Ks_u[buf][key_l * 33 + 16 + fp];   // feats 32..63
            }
            v8f t = __builtin_amdgcn_wmma_f32_16x16x32_bf16(
                false, aq[0].v, false, bk0.v, (short)0, vzero8(), false, false);
            s[a] = __builtin_amdgcn_wmma_f32_16x16x32_bf16(
                false, aq[1].v, false, bk1.v, (short)0, t, false, false);
        }

        // ---- online softmax (per C-row M = r + 8*half) ----
        bool ok[4];
#pragma unroll
        for (int a = 0; a < 4; a++) ok[a] = (kbase + a * 16 + lm) < len;
#pragma unroll
        for (int r = 0; r < 8; r++) {
            float sv[4];
            float rm = -1e30f;
#pragma unroll
            for (int a = 0; a < 4; a++) {
                sv[a] = ok[a] ? s[a][r] * scale : -1e30f;
                rm = fmaxf(rm, sv[a]);
            }
            rm = rowmax16(rm);
            float mn = fmaxf(mrow[r], rm);
            float corr = __expf(mrow[r] - mn);
            float rs = 0.0f;
            int Mr = r + 8 * half;
#pragma unroll
            for (int a = 0; a < 4; a++) {
                float p = __expf(sv[a] - mn);
                rs += p;
                p_lds[w][Mr * 65 + a * 16 + lm] = p;  // wave-internal: in-order
            }
            rs = rowsum16(rs);
            lrow[r] = lrow[r] * corr + rs;
            mrow[r] = mn;
#pragma unroll
            for (int a = 0; a < 4; a++) o[a][r] *= corr;
        }

        // ---- P (C layout) -> A layout via LDS, pack bf16; two K-chunks ----
        U16 ap[2];
#pragma unroll
        for (int fc = 0; fc < 2; fc++)
#pragma unroll
            for (int r = 0; r < 8; r++) {
                int kf = fc * 32 + ((r < 4) ? 0 : 16) + (half ? 8 : 0) + 2 * (r & 3);
                unsigned int b0 = f2bf(p_lds[w][lm * 65 + kf]);
                unsigned int b1 = f2bf(p_lds[w][lm * 65 + kf + 1]);
                ap[fc].u[r] = b0 | (b1 << 16);
            }

        // ---- O += P @ V : four 16-wide dv subtiles, key-dim 64 split 2x32 ----
#pragma unroll
        for (int a = 0; a < 4; a++) {
            int dv = a * 16 + lm;              // lane = B column N = dv
            U16 bv0, bv1;
#pragma unroll
            for (int r = 0; r < 8; r++) {
                int kp = ((r < 4) ? 0 : 8) + half * 4 + (r & 3);         // key-pair per B layout
                bv0.u[r] = Vs_u[buf][dv * 34 + kp];        // keys 0..31
                bv1.u[r] = Vs_u[buf][dv * 34 + 16 + kp];   // keys 32..63
            }
            o[a] = __builtin_amdgcn_wmma_f32_16x16x32_bf16(
                false, ap[0].v, false, bv0.v, (short)0, o[a], false, false);
            o[a] = __builtin_amdgcn_wmma_f32_16x16x32_bf16(
                false, ap[1].v, false, bv1.v, (short)0, o[a], false, false);
        }

        // ---- drain pipeline: store next V tile (other buffer), sync ----
        if (more) {
            unsigned short* Vs_s = (unsigned short*)Vs_u[buf ^ 1];
#pragma unroll
            for (int i = 0; i < 8; i++) {
                int d = tid + i * 256;
                int key = d >> 5, dp = d & 31;
                int dv0 = dp * 2;
                Vs_s[dv0 * 68 + key]       = (unsigned short)(vreg[i] & 0xFFFFu);
                Vs_s[(dv0 + 1) * 68 + key] = (unsigned short)(vreg[i] >> 16);
            }
            wait_async0();
        }
        __syncthreads();
    }

    // normalize and store (C layout)
#pragma unroll
    for (int a = 0; a < 4; a++) {
        int dv = a * 16 + lm;
#pragma unroll
        for (int r = 0; r < 8; r++) {
            int qrow = qbase + r + 8 * half;
            float val = o[a][r] / lrow[r];
            ao[(size_t)(b * SS + qrow) * (HH * DKV) + h * DKV + dv] = f2bf(val);
        }
    }
}

// ---------------- launch ----------------
extern "C" void kernel_launch(void* const* d_in, const int* in_sizes, int n_in,
                              void* d_out, int out_size, void* d_ws, size_t ws_size,
                              hipStream_t stream) {
    (void)in_sizes; (void)n_in; (void)out_size; (void)ws_size;

    const float* x    = (const float*)d_in[0];
    const int*   lens = (const int*)d_in[1];
    const float* ln_g = (const float*)d_in[2];
    const float* ln_b = (const float*)d_in[3];
    const float* Wq   = (const float*)d_in[4];
    const float* bq   = (const float*)d_in[5];
    const float* Wkv  = (const float*)d_in[6];
    const float* bkv  = (const float*)d_in[7];
    const float* Wo   = (const float*)d_in[8];
    const float* bo   = (const float*)d_in[9];
    float* out = (float*)d_out;

    char* ws = (char*)d_ws;
    size_t off = 0;
    auto alloc = [&](size_t bytes) -> void* {
        void* p = ws + off;
        off += (bytes + 255) & ~(size_t)255;
        return p;
    };
    const int MR = BB * SS;  // 8192 rows
    unsigned short* xn  = (unsigned short*)alloc((size_t)MR * DD * 2);
    unsigned short* qb  = (unsigned short*)alloc((size_t)MR * HH * DKV * 2);
    unsigned short* kvb = (unsigned short*)alloc((size_t)MR * 2 * HH * DKV * 2);
    unsigned short* aob = (unsigned short*)alloc((size_t)MR * HH * DKV * 2);
    unsigned short* wqb  = (unsigned short*)alloc((size_t)DD * HH * DKV * 2);
    unsigned short* wkvb = (unsigned short*)alloc((size_t)DD * 2 * HH * DKV * 2);
    unsigned short* wob  = (unsigned short*)alloc((size_t)HH * DKV * DD * 2);

    // 1) LayerNorm + cast
    ln_cast_kernel<<<MR, 256, 0, stream>>>(x, ln_g, ln_b, xn);

    // 2) weight casts
    cast_kernel<<<(DD * HH * DKV + 255) / 256, 256, 0, stream>>>(Wq, wqb, DD * HH * DKV);
    cast_kernel<<<(DD * 2 * HH * DKV + 255) / 256, 256, 0, stream>>>(Wkv, wkvb, DD * 2 * HH * DKV);
    cast_kernel<<<(HH * DKV * DD + 255) / 256, 256, 0, stream>>>(Wo, wob, HH * DKV * DD);

    // 3) projections: q = xn@Wq+bq ; kv = xn@Wkv+bkv
    gemm_bf16_kernel<false><<<dim3((HH * DKV) / 64, MR / 128), 256, 0, stream>>>(
        xn, wqb, bq, nullptr, qb, nullptr, MR, HH * DKV, DD);
    gemm_bf16_kernel<false><<<dim3((2 * HH * DKV) / 64, MR / 128), 256, 0, stream>>>(
        xn, wkvb, bkv, nullptr, kvb, nullptr, MR, 2 * HH * DKV, DD);

    // 4) flash attention
    attn_kernel<<<dim3(SS / 128, HH, BB), 256, 0, stream>>>(qb, kvb, lens, aob);

    // 5) output projection + residual
    gemm_bf16_kernel<true><<<dim3(DD / 64, MR / 128), 256, 0, stream>>>(
        aob, wob, bo, x, nullptr, out, MR, DD, DD);
}
